// SingleHeadSiameseAttention_30485677867891
// MI455X (gfx1250) — compile-verified, matrix operand order checked by reading, not run on hardware
//
#include <hip/hip_runtime.h>
#include <cstdint>

// Problem constants (B=8, LQ=LK=2048, D=1024)
#define B_    8
#define LSEQ_ 2048
#define D_    1024
#define RKE_  2080            // padded key rows per batch = 65*32 (row 2048 = dummy, 2049+ masked)
#define NROWS_ (B_ * LSEQ_)   // 16384 flattened rows

typedef __attribute__((ext_vector_type(16))) __bf16 v16bf;
typedef __attribute__((ext_vector_type(8)))  float  v8f;

union TileU { v16bf v; uint4 u[2]; };

// round-to-nearest-even f32 -> bf16
__device__ __forceinline__ unsigned short f2bf(float f) {
  union { float f; unsigned int u; } c; c.f = f;
  unsigned int u = c.u;
  u += 0x7FFFu + ((u >> 16) & 1u);
  return (unsigned short)(u >> 16);
}

// Load one WMMA bf16 operand tile (16 rows x 32 K) in the CDNA5 VGPR layout:
// lane r (0..15), hi = lane>>4: dwords 0-3 = K cols [k0+hi*8 .. +7],
// dwords 4-7 = K cols [k0+16+hi*8 .. +7]  => two 16B loads at +0 and +32 bytes.
// Works for A (rows = M) and for B when the matrix is stored row-major [N][K].
__device__ __forceinline__ v16bf load_tile(const unsigned short* base, int row, int ld, int k0, int hi) {
  const uint4* p = (const uint4*)(base + (size_t)row * ld + k0 + hi * 8);
  TileU t; t.u[0] = p[0]; t.u[1] = p[2];
  return t.v;
}

__device__ __forceinline__ v8f wmma_bf16(v16bf a, v16bf b, v8f c) {
  // (neg_a, A, neg_b, B, c_mod, C, reuse_a, reuse_b)
  return __builtin_amdgcn_wmma_f32_16x16x32_bf16(false, a, false, b, (short)0, c, false, false);
}

// ---------------------------------------------------------------------------
// Elementwise staging kernels
// ---------------------------------------------------------------------------
__global__ __launch_bounds__(256) void convert_kernel(const float* __restrict__ src,
                                                      unsigned short* __restrict__ dst, int n) {
  int i = blockIdx.x * 256 + threadIdx.x;
  if (i < n) dst[i] = f2bf(src[i]);
}

// kp pad rows 2048..2079 per batch: row 2048 = bf16(dummy), rest 0.
__global__ __launch_bounds__(256) void pad_kernel(const float* __restrict__ dummy,
                                                  unsigned short* __restrict__ kpb) {
  int idx = blockIdx.x * 256 + threadIdx.x;      // 8 * 32 * 1024 = 262144
  int b  = idx >> 15;
  int r  = (idx >> 10) & 31;
  int c  = idx & 1023;
  size_t off = ((size_t)b * RKE_ + 2048 + r) * D_ + c;
  kpb[off] = (r == 0) ? f2bf(dummy[c]) : (unsigned short)0;
}

// blockIdx.y selects: 0: q -> fused[:,1024:2048] (residual), 1: k -> kb.
__global__ __launch_bounds__(256) void stage_inputs_kernel(const float* __restrict__ q,
                                                           const float* __restrict__ k,
                                                           unsigned short* __restrict__ fusedb,
                                                           unsigned short* __restrict__ kb) {
  size_t idx = (size_t)blockIdx.x * 256 + threadIdx.x;   // < 16777216
  int row = (int)(idx >> 10);
  int col = (int)(idx & 1023);
  if (blockIdx.y == 0) fusedb[(size_t)row * 2048 + 1024 + col] = f2bf(q[idx]);
  else                 kb[idx] = f2bf(k[idx]);
}

// Tiled transpose+convert: v/tsp [B][key][d] f32 -> vT/tT [B][d][RKE] bf16
// (pad key rows 2048..2079 = 0). 32x32 tiles through LDS, coalesced both ways.
// grid = (2080 /*kt*32+dt*/, B, 2 /*v,tsp*/), block = 256.
__global__ __launch_bounds__(256) void transpose_kernel(const float* __restrict__ v,
                                                        const float* __restrict__ tsp,
                                                        unsigned short* __restrict__ vT,
                                                        unsigned short* __restrict__ tT) {
  __shared__ unsigned short sh[32][33];
  const int kt = blockIdx.x >> 5;       // 0..64  (key tile)
  const int dt = blockIdx.x & 31;       // 0..31  (d tile)
  const int b  = blockIdx.y;
  const float* src = blockIdx.z ? tsp : v;
  unsigned short* dst = blockIdx.z ? tT : vT;
  const int k0 = kt * 32, d0 = dt * 32;
  const int tx = threadIdx.x & 31;
  const int ty = threadIdx.x >> 5;      // 0..7
#pragma unroll
  for (int r = 0; r < 32; r += 8) {
    int key = k0 + ty + r;
    float val = (key < LSEQ_) ? src[((size_t)b * LSEQ_ + key) * D_ + d0 + tx] : 0.0f;
    sh[tx][ty + r] = f2bf(val);         // sh[d_local][key_local]
  }
  __syncthreads();
#pragma unroll
  for (int r = 0; r < 32; r += 8) {
    int d = d0 + ty + r;
    dst[((size_t)b * D_ + d) * RKE_ + k0 + tx] = sh[ty + r][tx];
  }
}

// ---------------------------------------------------------------------------
// Generic WMMA GEMM: C[M,N] = A[M,K](bf16, row-major) * Bw[N,K](bf16, row-major)^T
// Workgroup = 256 threads = 8 waves; each WG does 16 rows, wave w covers N cols
// [w*16*TN, w*16*TN + 16*TN). Per K-step all TN B tiles are preloaded before the
// WMMA chain so the loads batch into clauses and overlap the matrix pipe.
// MODE: 0 = bf16 store; 1 = +bias, relu, bf16 store; 2 = +bias, f32 store.
// ROWMAP: 1 remaps dest row r -> (r/2048)*RKE + r%2048.
// ---------------------------------------------------------------------------
template <int TN, int MODE, int ROWMAP>
__global__ __launch_bounds__(256) void gemm_bf16_kernel(const unsigned short* __restrict__ A, int lda,
                                                        const unsigned short* __restrict__ Bw, int ldb,
                                                        const float* __restrict__ bias,
                                                        void* __restrict__ Cout, int ldc, int K) {
  const int lane = threadIdx.x & 31;
  const int wave = threadIdx.x >> 5;
  const int lr = lane & 15, hi = lane >> 4;
  const int row0 = blockIdx.x * 16;
  const int n0 = wave * 16 * TN;

  v8f acc[TN];
#pragma unroll
  for (int t = 0; t < TN; ++t)
#pragma unroll
    for (int i = 0; i < 8; ++i) acc[t][i] = 0.0f;

  for (int k0 = 0; k0 < K; k0 += 32) {
    v16bf a = load_tile(A, row0 + lr, lda, k0, hi);
    v16bf bt[TN];
#pragma unroll
    for (int t = 0; t < TN; ++t) bt[t] = load_tile(Bw, n0 + t * 16 + lr, ldb, k0, hi);
#pragma unroll
    for (int t = 0; t < TN; ++t) acc[t] = wmma_bf16(a, bt[t], acc[t]);
  }

  // C layout: VGPR i -> row = i + 8*hi, col = n0 + t*16 + (lane&15)
#pragma unroll
  for (int t = 0; t < TN; ++t) {
#pragma unroll
    for (int i = 0; i < 8; ++i) {
      int r = row0 + i + 8 * hi;
      int c = n0 + t * 16 + lr;
      float v = acc[t][i];
      if (MODE >= 1) v += bias[c];
      if (MODE == 1) v = fmaxf(v, 0.0f);
      size_t drow = ROWMAP ? ((size_t)(r >> 11) * RKE_ + (size_t)(r & 2047)) : (size_t)r;
      if (MODE == 2) ((float*)Cout)[drow * (size_t)ldc + c] = v;
      else           ((unsigned short*)Cout)[drow * (size_t)ldc + c] = f2bf(v);
    }
  }
}

// ---------------------------------------------------------------------------
// Flash-style attention: per WG (16 q-rows of one batch), 8 waves each own a
// 128-wide D-slice. Stream 65 key tiles of 32 (2049 valid keys incl. dummy).
// Partial scores reduced across waves via ds_add_f32; wave 0 does the online
// softmax and publishes bf16 probabilities; PV reads pre-transposed V/tsp
// ([B][d][RKE] bf16) straight from global (L2-resident per batch) in the WMMA
// B layout -- no in-loop staging, only ~3.3 KB LDS per workgroup.
// Epilogue: x1 = out*res, x2 = res-out (bf16 -> ws), out_tsp (f32 -> d_out).
// ---------------------------------------------------------------------------
__global__ __launch_bounds__(256) void attn_kernel(const unsigned short* __restrict__ qp,
                                                   const unsigned short* __restrict__ kp,
                                                   const unsigned short* __restrict__ vT,
                                                   const unsigned short* __restrict__ tT,
                                                   const float* __restrict__ qres,
                                                   float* __restrict__ out_tsp,
                                                   unsigned short* __restrict__ x1b,
                                                   unsigned short* __restrict__ x2b) {
  __shared__ float scoreBuf[16 * 32];
  __shared__ __attribute__((aligned(16))) unsigned short pBuf[16 * 32];
  __shared__ float alphaBuf[16];
  __shared__ float mBuf[16];
  __shared__ float lBuf[16];

  const int lane = threadIdx.x & 31;
  const int wave = threadIdx.x >> 5;
  const int lr = lane & 15, hi = lane >> 4;
  const int b  = blockIdx.x >> 7;     // 128 q-tiles per batch
  const int qt = blockIdx.x & 127;
  const int qrow0 = qt * 16;
  const int d0 = wave * 128;

  const unsigned short* qpt   = qp + ((size_t)b * LSEQ_ + qrow0) * D_;
  const unsigned short* kpb_b = kp + (size_t)b * RKE_ * D_;
  const unsigned short* vTb   = vT + (size_t)b * D_ * RKE_;
  const unsigned short* tTb   = tT + (size_t)b * D_ * RKE_;

  // Q operands for this wave's 128-dim slice (4 K-steps of 32), kept in VGPRs.
  v16bf aq[4];
#pragma unroll
  for (int s = 0; s < 4; ++s) aq[s] = load_tile(qpt, lr, D_, d0 + s * 32, hi);

  v8f accv[8], acct[8];
#pragma unroll
  for (int t = 0; t < 8; ++t)
#pragma unroll
    for (int i = 0; i < 8; ++i) { accv[t][i] = 0.0f; acct[t][i] = 0.0f; }

  if (wave == 0 && lane < 16) { mBuf[lane] = -INFINITY; lBuf[lane] = 0.0f; }

  for (int it = 0; it < 65; ++it) {
    const int kbase = it * 32;
    __syncthreads();
    scoreBuf[threadIdx.x] = 0.0f;
    scoreBuf[threadIdx.x + 256] = 0.0f;
    __syncthreads();

    // L2 prefetch of next key tile for this wave's slice
    if (kbase + 32 < RKE_)
      __builtin_prefetch(kpb_b + (size_t)(kbase + 32 + lr) * D_ + d0, 0, 1);

    // partial scores: this wave's 128-dim contribution to the 16x32 score tile.
    // The 4 B tiles of each 16-key half are loaded up front so the clause engine can batch them.
#pragma unroll
    for (int j = 0; j < 2; ++j) {
      v16bf bk[4];
#pragma unroll
      for (int st = 0; st < 4; ++st)
        bk[st] = load_tile(kpb_b, kbase + j * 16 + lr, D_, d0 + st * 32, hi);
      v8f s;
#pragma unroll
      for (int i = 0; i < 8; ++i) s[i] = 0.0f;
#pragma unroll
      for (int st = 0; st < 4; ++st) s = wmma_bf16(aq[st], bk[st], s);
#pragma unroll
      for (int i = 0; i < 8; ++i)
        atomicAdd(&scoreBuf[(i + 8 * hi) * 32 + j * 16 + lr], s[i]);
    }
    __syncthreads();

    // online softmax on wave 0 (lanes 0..15 own one row each)
    if (wave == 0 && lane < 16) {
      const int r = lane;
      float mold = mBuf[r], lold = lBuf[r];
      float sv[32];
      float mnew = mold;
#pragma unroll
      for (int c = 0; c < 32; ++c) {
        float sc = scoreBuf[r * 32 + c] * 0.03125f;   // 1/sqrt(1024)
        if (kbase + c >= 2049) sc = -INFINITY;        // mask pad keys
        sv[c] = sc;
        mnew = fmaxf(mnew, sc);
      }
      float alpha = __expf(mold - mnew);
      float lsum = 0.0f;
#pragma unroll
      for (int c = 0; c < 32; c += 2) {
        float p0 = __expf(sv[c] - mnew);
        float p1 = __expf(sv[c + 1] - mnew);
        lsum += p0 + p1;
        unsigned int pk = (unsigned int)f2bf(p0) | ((unsigned int)f2bf(p1) << 16);
        *(unsigned int*)(pBuf + r * 32 + c) = pk;
      }
      mBuf[r] = mnew;
      lBuf[r] = alpha * lold + lsum;
      alphaBuf[r] = alpha;
    }
    __syncthreads();

    // rescale accumulators by per-row alpha
    float al[8];
#pragma unroll
    for (int i = 0; i < 8; ++i) al[i] = alphaBuf[i + 8 * hi];
#pragma unroll
    for (int t = 0; t < 8; ++t)
#pragma unroll
      for (int i = 0; i < 8; ++i) { accv[t][i] *= al[i]; acct[t][i] *= al[i]; }

    // probability A-operand (16 q-rows x 32 keys) from LDS
    v16bf ap = load_tile(pBuf, lr, 32, 0, hi);

    // PV: B operands come straight from the pre-transposed global buffers
    {
      v16bf bv[8];
#pragma unroll
      for (int t = 0; t < 8; ++t)
        bv[t] = load_tile(vTb, d0 + t * 16 + lr, RKE_, kbase, hi);
#pragma unroll
      for (int t = 0; t < 8; ++t) accv[t] = wmma_bf16(ap, bv[t], accv[t]);
#pragma unroll
      for (int t = 0; t < 8; ++t)
        bv[t] = load_tile(tTb, d0 + t * 16 + lr, RKE_, kbase, hi);
#pragma unroll
      for (int t = 0; t < 8; ++t) acct[t] = wmma_bf16(ap, bv[t], acct[t]);
    }
  }
  __syncthreads();

  // epilogue: normalize, fuse with residual, emit x1/x2 (bf16) and out_tsp (f32)
  float linv[8];
#pragma unroll
  for (int i = 0; i < 8; ++i) linv[i] = 1.0f / lBuf[i + 8 * hi];

  const float* qsrc = qres + ((size_t)b * LSEQ_ + qrow0) * D_;
#pragma unroll
  for (int t = 0; t < 8; ++t) {
#pragma unroll
    for (int i = 0; i < 8; ++i) {
      int row = i + 8 * hi;
      int col = d0 + t * 16 + lr;
      float ov = accv[t][i] * linv[i];
      float ot = acct[t][i] * linv[i];
      float rs = qsrc[(size_t)row * D_ + col];
      size_t gidx = ((size_t)b * LSEQ_ + qrow0 + row) * D_ + col;
      x1b[gidx] = f2bf(ov * rs);
      x2b[gidx] = f2bf(rs - ov);
      out_tsp[gidx] = ot;
    }
  }
}

// ---------------------------------------------------------------------------
// Workspace layout (bytes). Total ~312 MB.
// ---------------------------------------------------------------------------
static constexpr size_t QP_OFF    = 0;                               // 16384x1024 bf16
static constexpr size_t KP_OFF    = QP_OFF    + 33554432ull;         // 8x2080x1024 bf16
static constexpr size_t VT_OFF    = KP_OFF    + 34078720ull;         // 8x1024x2080 bf16 (transposed)
static constexpr size_t TT_OFF    = VT_OFF    + 34078720ull;
static constexpr size_t KB_OFF    = TT_OFF    + 34078720ull;         // 16384x1024 bf16
static constexpr size_t FUSED_OFF = KB_OFF    + 33554432ull;         // 16384x2048 bf16
static constexpr size_t X1_OFF    = FUSED_OFF + 67108864ull;
static constexpr size_t X2_OFF    = X1_OFF    + 33554432ull;
static constexpr size_t WQK_OFF   = X2_OFF    + 33554432ull;         // 1024x1024 bf16
static constexpr size_t W1_OFF    = WQK_OFF   + 2097152ull;          // 512x1024 bf16
static constexpr size_t W2_OFF    = W1_OFF    + 1048576ull;
static constexpr size_t W3_OFF    = W2_OFF    + 1048576ull;          // 1024x2048 bf16

extern "C" void kernel_launch(void* const* d_in, const int* in_sizes, int n_in,
                              void* d_out, int out_size, void* d_ws, size_t ws_size,
                              hipStream_t stream) {
  (void)in_sizes; (void)n_in; (void)out_size; (void)ws_size;

  const float* q     = (const float*)d_in[0];
  const float* k     = (const float*)d_in[1];
  const float* v     = (const float*)d_in[2];
  const float* tsp   = (const float*)d_in[3];
  const float* w_qk  = (const float*)d_in[4];
  const float* dummy = (const float*)d_in[5];
  const float* w1    = (const float*)d_in[6];
  const float* b1    = (const float*)d_in[7];
  const float* w2    = (const float*)d_in[8];
  const float* b2    = (const float*)d_in[9];
  const float* w3    = (const float*)d_in[10];
  const float* b3    = (const float*)d_in[11];

  char* ws = (char*)d_ws;
  unsigned short* qpb    = (unsigned short*)(ws + QP_OFF);
  unsigned short* kpb    = (unsigned short*)(ws + KP_OFF);
  unsigned short* vTb    = (unsigned short*)(ws + VT_OFF);
  unsigned short* tTb    = (unsigned short*)(ws + TT_OFF);
  unsigned short* kb     = (unsigned short*)(ws + KB_OFF);
  unsigned short* fusedb = (unsigned short*)(ws + FUSED_OFF);
  unsigned short* x1b    = (unsigned short*)(ws + X1_OFF);
  unsigned short* x2b    = (unsigned short*)(ws + X2_OFF);
  unsigned short* wqkb   = (unsigned short*)(ws + WQK_OFF);
  unsigned short* w1b    = (unsigned short*)(ws + W1_OFF);
  unsigned short* w2b    = (unsigned short*)(ws + W2_OFF);
  unsigned short* w3b    = (unsigned short*)(ws + W3_OFF);

  // 1) weights -> bf16
  convert_kernel<<<(1048576 + 255) / 256, 256, 0, stream>>>(w_qk, wqkb, 1048576);
  convert_kernel<<<(524288 + 255) / 256, 256, 0, stream>>>(w1, w1b, 524288);
  convert_kernel<<<(524288 + 255) / 256, 256, 0, stream>>>(w2, w2b, 524288);
  convert_kernel<<<(2097152 + 255) / 256, 256, 0, stream>>>(w3, w3b, 2097152);

  // 2) dummy key row + zero pad rows of kp
  pad_kernel<<<1024, 256, 0, stream>>>(dummy, kpb);

  // 3) q -> fused residual cols, k -> kb; v/tsp -> transposed bf16 [B][d][RKE]
  stage_inputs_kernel<<<dim3(65536, 2, 1), 256, 0, stream>>>(q, k, fusedb, kb);
  transpose_kernel<<<dim3(2080, B_, 2), 256, 0, stream>>>(v, tsp, vTb, tTb);

  // 4) qp = q @ Wqk^T, kp = k @ Wqk^T (kp with per-batch row remap into padded buffer)
  gemm_bf16_kernel<8, 0, 0><<<1024, 256, 0, stream>>>(fusedb + 1024, 2048, wqkb, 1024, nullptr,
                                                      (void*)qpb, 1024, 1024);
  gemm_bf16_kernel<8, 0, 1><<<1024, 256, 0, stream>>>(kb, 1024, wqkb, 1024, nullptr,
                                                      (void*)kpb, 1024, 1024);

  // 5) attention (writes x1,x2 bf16 and out_tsp f32 = second output)
  attn_kernel<<<dim3(B_ * 128, 1, 1), 256, 0, stream>>>(qpb, kpb, vTb, tTb, q,
                                                        (float*)d_out + (size_t)NROWS_ * D_,
                                                        x1b, x2b);

  // 6) o1, o2 into fused cols [0,512) and [512,1024)
  gemm_bf16_kernel<4, 1, 0><<<1024, 256, 0, stream>>>(x1b, 1024, w1b, 1024, b1,
                                                      (void*)fusedb, 2048, 1024);
  gemm_bf16_kernel<4, 1, 0><<<1024, 256, 0, stream>>>(x2b, 1024, w2b, 1024, b2,
                                                      (void*)(fusedb + 512), 2048, 1024);

  // 7) output = fused @ W3^T + b3 (f32 -> first output)
  gemm_bf16_kernel<8, 2, 0><<<1024, 256, 0, stream>>>(fusedb, 2048, w3b, 2048, b3,
                                                      d_out, 1024, 2048);
}
